// AttentionRNNClassifer_25915832664089
// MI455X (gfx1250) — matrix operand order repaired and checked by reading, not compiled, not run on hardware
//
#include <hip/hip_runtime.h>
#include <hip/hip_bf16.h>
#include <math.h>

// Problem constants (from reference): B=32, T=128, E=256, H=256, C=16
#define B_  32
#define T_  128
#define E_  256
#define H_  256
#define C_  16
#define H3_ 768

typedef __attribute__((ext_vector_type(16))) _Float16 v16h;
typedef __attribute__((ext_vector_type(8)))  float    v8f;

__device__ __forceinline__ float sigmoidf_(float x) {
  return 1.0f / (1.0f + __expf(-x));
}

// One 16x16 output tile, K-reduction in steps of 32 via v_wmma_f32_16x16x32_f16.
// arow = &A[m0+lr][0] (A row-major, stride = K), brow = &W[n0+lr][0] (W row-major [N][K]).
// CDNA5 A layout (16-bit 16x32): lane L holds row M=L&15; VGPR v holds
//   K = (v<4 ? 2v : 16+2(v-4)) + (L>=16 ? 8 : 0), pairs (K,K+1).
// B layout (16-bit 32x16): lane L holds col N=L&15; half j holds K = (j<8 ? j : 8+j) + (L>=16 ? 8 : 0).
__device__ __forceinline__ v8f wmma_acc_16x16(const _Float16* __restrict__ arow,
                                              const _Float16* __restrict__ brow,
                                              int hi, int K, v8f acc) {
  for (int k0 = 0; k0 < K; k0 += 32) {
    v16h a, b;
#pragma unroll
    for (int v = 0; v < 8; ++v) {
      int kb = k0 + ((v < 4) ? (2 * v) : (16 + 2 * (v - 4))) + hi * 8;
      a[2 * v]     = arow[kb];
      a[2 * v + 1] = arow[kb + 1];
    }
#pragma unroll
    for (int j = 0; j < 16; ++j) b[j] = brow[k0 + hi * 8 + j + (j >= 8 ? 8 : 0)];
    acc = __builtin_amdgcn_wmma_f32_16x16x32_f16(false, a, false, b,
                                                 (short)0, acc, false, false);
  }
  return acc;
}

// ---------------------------------------------------------------------------
__global__ void cvt_f32_f16_kernel(const float* __restrict__ s,
                                   _Float16* __restrict__ d, int n) {
  int i = blockIdx.x * blockDim.x + threadIdx.x;
  if (i < n) d[i] = (_Float16)s[i];
}

__global__ void init_kernel(float* __restrict__ hlast32,
                            _Float16* __restrict__ hlast16,
                            int* __restrict__ flags) {
  int i = blockIdx.x * blockDim.x + threadIdx.x;
  if (i < B_ * H_) { hlast32[i] = 0.f; hlast16[i] = (_Float16)0.f; }
  if (i <= T_) flags[i] = (i == 0) ? 1 : 0;
}

// xp = x @ W_in.T + b_in, stored TIME-MAJOR [T][B][H] in f32 and f16.
// x16 rows are (b*T + t), output rows are (t*B + b).
__global__ void xp_gemm_kernel(const _Float16* __restrict__ x16,
                               const _Float16* __restrict__ Win16,
                               const float* __restrict__ b_in,
                               float* __restrict__ xp32,
                               _Float16* __restrict__ xp16) {
  const int tid = threadIdx.x, lane = tid & 31, lr = lane & 15, hi = lane >> 4;
  const int wave = blockIdx.x * 8 + (tid >> 5);
  const int tilesM = (B_ * T_) / 16;     // 256
  const int tn = wave / tilesM;          // 0..15
  const int tm = wave % tilesM;
  if (tn >= H_ / 16) return;
  const int m0 = tm * 16, n0 = tn * 16;
  v8f acc = {};
  acc = wmma_acc_16x16(x16 + (size_t)(m0 + lr) * E_,
                       Win16 + (size_t)(n0 + lr) * E_, hi, E_, acc);
  const int col = n0 + lr;
  const float bv = b_in[col];
#pragma unroll
  for (int r = 0; r < 8; ++r) {
    const int g = m0 + r + 8 * hi;       // row in [B*T]
    const int b = g / T_, t = g % T_;
    float v = acc[r] + bv;
    size_t o = ((size_t)t * B_ + b) * H_ + col;
    xp32[o] = v;
    xp16[o] = (_Float16)v;
  }
}

// Generic Y = A @ W.T + bias  (A: [M][K] f16 row-major, W: [N][K] f16 row-major)
__global__ void gemm_bias_kernel(const _Float16* __restrict__ A, int lda,
                                 const _Float16* __restrict__ W, int ldw,
                                 const float* __restrict__ bias,
                                 float* __restrict__ Y, int ldy,
                                 int M, int N) {
  const int tid = threadIdx.x, lane = tid & 31, lr = lane & 15, hi = lane >> 4;
  const int wave = blockIdx.x * 8 + (tid >> 5);
  const int tilesM = M >> 4, tilesN = N >> 4;
  if (wave >= tilesM * tilesN) return;
  const int tm = wave % tilesM, tn = wave / tilesM;
  const int m0 = tm * 16, n0 = tn * 16;
  v8f acc = {};
  acc = wmma_acc_16x16(A + (size_t)(m0 + lr) * lda,
                       W + (size_t)(n0 + lr) * ldw, hi, lda, acc);
  const int col = n0 + lr;
  const float bv = bias[col];
#pragma unroll
  for (int r = 0; r < 8; ++r)
    Y[(size_t)(m0 + r + 8 * hi) * ldy + col] = acc[r] + bv;
}

// ---------------------------------------------------------------------------
// Main sequential kernel: block i == outer step i.  Daisy-chained via flags[].
// 8 waves/block; wave w owns hidden columns [32w, 32w+32).
__global__ __launch_bounds__(256) void attn_rnn_seq_kernel(
    const _Float16* __restrict__ Whh16, const float* __restrict__ bhh,
    const _Float16* __restrict__ Wa16,  const float* __restrict__ ba,
    const float* __restrict__ va_w,     const float* __restrict__ va_b,
    const _Float16* __restrict__ Wo16,  const float* __restrict__ bo,
    const float* __restrict__ xw,       // [T][B][3H] (includes bih)
    const float* __restrict__ uax,      // [T][B][H]  (includes ub)
    const float* __restrict__ xp32,     // [T][B][H]
    float* __restrict__ hlast32,        // [T+1][B][H]
    _Float16* __restrict__ hlast16,     // [T+1][B][H]
    int* __restrict__ flags,            // [T+1]
    float* __restrict__ out)            // [B][T][C]
{
  __shared__ _Float16 h16[2][B_ * H_];   // ping-pong f16 hidden (A operand)
  __shared__ float    h32[B_ * H_];      // f32 hidden (owner-exclusive r/w)
  __shared__ float    ctx[B_ * H_];      // online-softmax context accumulator
  __shared__ float    score_sh[B_];
  __shared__ float    mrun[B_], lrun[B_], scl[B_], ew[B_];

  const int i    = blockIdx.x;
  const int tid  = threadIdx.x;
  const int wave = tid >> 5, lane = tid & 31, lr = lane & 15, hi = lane >> 4;

  // Wait for h_last[i] from block i-1 (block 0 pre-released by init kernel).
  if (tid == 0) {
    while (__atomic_load_n(flags + i, __ATOMIC_ACQUIRE) == 0)
      __builtin_amdgcn_s_sleep(1);
  }
  __syncthreads();

  for (int idx = tid; idx < B_ * H_; idx += 256) {
    float v = hlast32[(size_t)i * B_ * H_ + idx];
    h32[idx]     = v;
    h16[0][idx]  = (_Float16)v;
    ctx[idx]     = 0.f;
  }
  if (tid < B_) { mrun[tid] = -1e30f; lrun[tid] = 0.f; }
  __syncthreads();

  int cur = 0;
  for (int t = 0; t <= i; ++t) {
    const int nxt = cur ^ 1;
    if (tid < B_) score_sh[tid] = 0.f;

    // ---- GRU step: hw = h @ Whh.T, gates fused in accumulator fragments ----
    const size_t xwbase = (size_t)t * B_ * H3_;
#pragma unroll
    for (int sub = 0; sub < 4; ++sub) {
      const int m0   = 16 * (sub & 1);
      const int col0 = 32 * wave + 16 * (sub >> 1);
      const _Float16* arow = &h16[cur][(m0 + lr) * H_];
      v8f ar = {}, az = {}, an = {};
      ar = wmma_acc_16x16(arow, Whh16 + (size_t)(0 * H_ + col0 + lr) * H_, hi, H_, ar);
      az = wmma_acc_16x16(arow, Whh16 + (size_t)(1 * H_ + col0 + lr) * H_, hi, H_, az);
      an = wmma_acc_16x16(arow, Whh16 + (size_t)(2 * H_ + col0 + lr) * H_, hi, H_, an);
      const int col = col0 + lr;
      const float bh_r = bhh[col], bh_z = bhh[H_ + col], bh_n = bhh[2 * H_ + col];
#pragma unroll
      for (int r = 0; r < 8; ++r) {
        const int row = m0 + r + 8 * hi;                 // batch index
        const size_t xb = xwbase + (size_t)row * H3_;
        float xr = xw[xb + col], xz = xw[xb + H_ + col], xn = xw[xb + 2 * H_ + col];
        float rg = sigmoidf_(xr + ar[r] + bh_r);
        float zg = sigmoidf_(xz + az[r] + bh_z);
        float ng = tanhf(xn + rg * (an[r] + bh_n));
        float ho = h32[row * H_ + col];
        float hn = (1.f - zg) * ng + zg * ho;
        h32[row * H_ + col]      = hn;
        h16[nxt][row * H_ + col] = (_Float16)hn;
      }
    }
    __syncthreads();   // (a) h_new visible

    // Publish chain state as early as possible (overlaps attention below).
    if (t == i) {
      for (int idx = tid; idx < B_ * H_; idx += 256) {
        hlast32[(size_t)(i + 1) * B_ * H_ + idx] = h32[idx];
        hlast16[(size_t)(i + 1) * B_ * H_ + idx] = h16[nxt][idx];
      }
      __threadfence();
    }

    // ---- attention score: tanh(h_new @ Wa.T + ba + uax[t]) . va_w ----
    const size_t uaxbase = (size_t)t * B_ * H_;
#pragma unroll
    for (int sub = 0; sub < 4; ++sub) {
      const int m0   = 16 * (sub & 1);
      const int col0 = 32 * wave + 16 * (sub >> 1);
      v8f as = {};
      as = wmma_acc_16x16(&h16[nxt][(m0 + lr) * H_],
                          Wa16 + (size_t)(col0 + lr) * H_, hi, H_, as);
      const int col  = col0 + lr;
      const float vw = va_w[col], bav = ba[col];
#pragma unroll
      for (int r = 0; r < 8; ++r) {
        const int row = m0 + r + 8 * hi;
        float sv  = tanhf(as[r] + bav + uax[uaxbase + (size_t)row * H_ + col]);
        float val = sv * vw;
        val += __shfl_xor(val, 1, 32);
        val += __shfl_xor(val, 2, 32);
        val += __shfl_xor(val, 4, 32);
        val += __shfl_xor(val, 8, 32);     // reduce over 16-lane half (fixed row)
        if (lr == 0) atomicAdd(&score_sh[row], val);
      }
    }
    __syncthreads();   // (b) scores complete, hlast stores fenced

    if (t == i && tid == 0)
      __atomic_store_n(flags + i + 1, 1, __ATOMIC_RELEASE);

    // ---- online softmax stats (per batch row) ----
    if (tid < B_) {
      float sc = score_sh[tid] + va_b[0];
      float mo = mrun[tid];
      float mn = fmaxf(mo, sc);
      float e  = __expf(sc - mn);
      float s  = __expf(mo - mn);
      lrun[tid] = lrun[tid] * s + e;
      mrun[tid] = mn;
      scl[tid]  = s;
      ew[tid]   = e;
    }
    __syncthreads();   // (c)

    // ---- streaming context accumulation: ctx = ctx*scl + e * xp[t] ----
    const size_t xpbase = (size_t)t * B_ * H_;
    for (int idx = tid; idx < B_ * H_; idx += 256) {
      int b = idx >> 8;                      // H_ == 256
      ctx[idx] = ctx[idx] * scl[b] + ew[b] * xp32[xpbase + idx];
    }
    __syncthreads();   // (d)

    cur = nxt;
  }

  // ---- logits = [h, ctx/l] @ Wo.T + bo ; out = sigmoid(logits) ----
  if (wave < 2) {
    const int m0 = wave * 16;
    const int arowi = m0 + lr;               // A-operand batch row (per lane)
    const float invl = 1.f / lrun[arowi];
    v8f acc = {};
    for (int k0 = 0; k0 < 2 * H_; k0 += 32) {
      v16h a, b;
#pragma unroll
      for (int v = 0; v < 8; ++v) {
        int kb = k0 + ((v < 4) ? (2 * v) : (16 + 2 * (v - 4))) + hi * 8;
#pragma unroll
        for (int u = 0; u < 2; ++u) {
          int kk = kb + u;
          a[2 * v + u] = (kk < H_)
              ? h16[cur][arowi * H_ + kk]
              : (_Float16)(ctx[arowi * H_ + (kk - H_)] * invl);
        }
      }
#pragma unroll
      for (int j = 0; j < 16; ++j) b[j] = Wo16[(size_t)lr * (2 * H_) + k0 + hi * 8 + j + (j >= 8 ? 8 : 0)];
      acc = __builtin_amdgcn_wmma_f32_16x16x32_f16(false, a, false, b,
                                                   (short)0, acc, false, false);
    }
#pragma unroll
    for (int r = 0; r < 8; ++r) {
      const int row = m0 + r + 8 * hi;       // batch index
      float lg = acc[r] + bo[lr];
      out[(size_t)row * T_ * C_ + (size_t)i * C_ + lr] = sigmoidf_(lg);
    }
  }
}

// ---------------------------------------------------------------------------
extern "C" void kernel_launch(void* const* d_in, const int* in_sizes, int n_in,
                              void* d_out, int out_size, void* d_ws, size_t ws_size,
                              hipStream_t stream) {
  (void)in_sizes; (void)n_in; (void)out_size; (void)ws_size;
  const float* x    = (const float*)d_in[0];
  const float* W_in = (const float*)d_in[1];
  const float* b_in = (const float*)d_in[2];
  const float* Wih  = (const float*)d_in[3];
  const float* bih  = (const float*)d_in[4];
  const float* Whh  = (const float*)d_in[5];
  const float* bhh  = (const float*)d_in[6];
  const float* Wa   = (const float*)d_in[7];
  const float* ba   = (const float*)d_in[8];
  const float* Ua   = (const float*)d_in[9];
  const float* ub   = (const float*)d_in[10];
  const float* va_w = (const float*)d_in[11];
  const float* va_b = (const float*)d_in[12];
  const float* Wo   = (const float*)d_in[13];
  const float* bo   = (const float*)d_in[14];
  float* out = (float*)d_out;

  char* p = (char*)d_ws;
  auto take = [&](size_t bytes) -> char* {
    char* r = p;
    p += (bytes + 255) & ~(size_t)255;
    return r;
  };

  _Float16* x16     = (_Float16*)take((size_t)B_ * T_ * E_ * sizeof(_Float16));
  _Float16* Win16   = (_Float16*)take((size_t)H_ * E_ * sizeof(_Float16));
  _Float16* Wih16   = (_Float16*)take((size_t)H3_ * H_ * sizeof(_Float16));
  _Float16* Whh16   = (_Float16*)take((size_t)H3_ * H_ * sizeof(_Float16));
  _Float16* Ua16    = (_Float16*)take((size_t)H_ * H_ * sizeof(_Float16));
  _Float16* Wa16    = (_Float16*)take((size_t)H_ * H_ * sizeof(_Float16));
  _Float16* Wo16    = (_Float16*)take((size_t)C_ * 2 * H_ * sizeof(_Float16));
  _Float16* xp16    = (_Float16*)take((size_t)T_ * B_ * H_ * sizeof(_Float16));
  float*    xp32    = (float*)take((size_t)T_ * B_ * H_ * sizeof(float));
  float*    xw      = (float*)take((size_t)T_ * B_ * H3_ * sizeof(float));
  float*    uax     = (float*)take((size_t)T_ * B_ * H_ * sizeof(float));
  float*    hlast32 = (float*)take((size_t)(T_ + 1) * B_ * H_ * sizeof(float));
  _Float16* hlast16 = (_Float16*)take((size_t)(T_ + 1) * B_ * H_ * sizeof(_Float16));
  int*      flags   = (int*)take((size_t)(T_ + 1) * sizeof(int));

  auto cvt = [&](const float* s, _Float16* d, int n) {
    cvt_f32_f16_kernel<<<(n + 255) / 256, 256, 0, stream>>>(s, d, n);
  };
  cvt(x,    x16,   B_ * T_ * E_);
  cvt(W_in, Win16, H_ * E_);
  cvt(Wih,  Wih16, H3_ * H_);
  cvt(Whh,  Whh16, H3_ * H_);
  cvt(Ua,   Ua16,  H_ * H_);
  cvt(Wa,   Wa16,  H_ * H_);
  cvt(Wo,   Wo16,  C_ * 2 * H_);

  init_kernel<<<(B_ * H_ + 255) / 256, 256, 0, stream>>>(hlast32, hlast16, flags);

  // xp = x @ W_in.T + b_in  (time-major)
  {
    int tiles = ((B_ * T_) / 16) * (H_ / 16);       // 4096
    xp_gemm_kernel<<<tiles / 8, 256, 0, stream>>>(x16, Win16, b_in, xp32, xp16);
  }
  // xw = xp @ Wih.T + bih   [T*B, 768]
  {
    int tiles = ((T_ * B_) / 16) * (H3_ / 16);      // 12288
    gemm_bias_kernel<<<(tiles + 7) / 8, 256, 0, stream>>>(
        xp16, H_, Wih16, H_, bih, xw, H3_, T_ * B_, H3_);
  }
  // uax = xp @ Ua.T + ub    [T*B, 256]
  {
    int tiles = ((T_ * B_) / 16) * (H_ / 16);       // 4096
    gemm_bias_kernel<<<(tiles + 7) / 8, 256, 0, stream>>>(
        xp16, H_, Ua16, H_, ub, uax, H_, T_ * B_, H_);
  }

  attn_rnn_seq_kernel<<<T_, 256, 0, stream>>>(
      Whh16, bhh, Wa16, ba, va_w, va_b, Wo16, bo,
      xw, uax, xp32, hlast32, hlast16, flags, out);
}
